// GNN_Model_26603027432073
// MI455X (gfx1250) — matrix operand order repaired and checked by reading, not compiled, not run on hardware
//
#include <hip/hip_runtime.h>
#include <math.h>

typedef __attribute__((ext_vector_type(16))) __bf16 v16bf;
typedef __attribute__((ext_vector_type(8)))  float  v8f;

struct BitPack32 { uint4 a, b; };   // 32 bytes == 16 bf16

__device__ __forceinline__ unsigned short f2bf(float f) {
  unsigned int u = __float_as_uint(f);
  u += 0x7FFFu + ((u >> 16) & 1u);          // round-to-nearest-even
  return (unsigned short)(u >> 16);
}

__device__ __forceinline__ void atomicMaxF(float* addr, float val) {
  unsigned int* ua = (unsigned int*)addr;
  unsigned int old = *ua;
  while (__uint_as_float(old) < val) {
    unsigned int assumed = old;
    old = atomicCAS(ua, assumed, __float_as_uint(val));
    if (old == assumed) break;
  }
}

__global__ void fill_f32(float* p, float v, int n) {
  int i = blockIdx.x * blockDim.x + threadIdx.x;
  if (i < n) p[i] = v;
}

__global__ void cvt_f32_bf16(const float* __restrict__ in, unsigned short* __restrict__ out, int n) {
  int i = blockIdx.x * blockDim.x + threadIdx.x;
  if (i < n) out[i] = f2bf(in[i]);
}

// W[K][M] (row-major f32) -> Wt[M][K] (row-major bf16) so B-operand loads are contiguous.
__global__ void cvt_w_t_bf16(const float* __restrict__ W, unsigned short* __restrict__ Wt,
                             int K, int M, int mShift) {
  int i = blockIdx.x * blockDim.x + threadIdx.x;
  if (i >= K * M) return;
  int k = i >> mShift;
  int m = i & (M - 1);
  Wt[(size_t)m * K + k] = f2bf(W[i]);
}

// H[N x Mout] = Xb[N x K](bf16) * Wt[Mout x K]^T(bf16), f32 accumulate.
// 1 wave = one 16x16 output tile; 4 waves/block over row tiles.
__global__ void gemm_bf16_wmma(const unsigned short* __restrict__ Xb,
                               const unsigned short* __restrict__ Wt,
                               float* __restrict__ H,
                               int nRowTiles, int K, int Mout) {
  int lane = threadIdx.x & 31;
  int wave = threadIdx.x >> 5;
  int rowTile = blockIdx.x * 4 + wave;
  if (rowTile >= nRowTiles) return;           // wave-uniform: EXEC all-ones for WMMA
  int hf  = lane >> 4;
  int l15 = lane & 15;
  const uint4* Arow = (const uint4*)(Xb + (size_t)(rowTile * 16 + l15) * K);
  const uint4* Brow = (const uint4*)(Wt + (size_t)(blockIdx.y * 16 + l15) * K);
  v8f acc = {0.f, 0.f, 0.f, 0.f, 0.f, 0.f, 0.f, 0.f};
  for (int kk = 0; kk < K; kk += 32) {
    int base = kk >> 3;                        // uint4 (8 x bf16) granularity
    BitPack32 ab, bb;
    ab.a = Arow[base + hf];                    // A: K = kk+8*hf .. +7
    ab.b = Arow[base + 2 + hf];                // A: K = kk+16+8*hf .. +7
    bb.a = Brow[base + hf];                    // B: K = kk+8*hf .. +7
    bb.b = Brow[base + 2 + hf];                // B: K = kk+16+8*hf .. +7
    v16bf va = __builtin_bit_cast(v16bf, ab);
    v16bf vb = __builtin_bit_cast(v16bf, bb);
    acc = __builtin_amdgcn_wmma_f32_16x16x32_bf16(false, va, false, vb,
                                                  (short)0, acc, false, false);
  }
  float* outp = H + (size_t)(rowTile * 16 + hf * 8) * Mout + blockIdx.y * 16 + l15;
#pragma unroll
  for (int i = 0; i < 8; ++i) outp[(size_t)i * Mout] = acc[i];
}

// a_src[n,h] = <h[n,h,:], att_src[h,:]>, same for dst. One thread per (n,head).
__global__ void attn_node(const float* __restrict__ h, const float* __restrict__ attS,
                          const float* __restrict__ attD, float* __restrict__ asrc,
                          float* __restrict__ adst, int Nr, int c) {
  int idx = blockIdx.x * blockDim.x + threadIdx.x;
  if (idx >= Nr * 2) return;
  int n = idx >> 1, head = idx & 1;
  const float* hp = h + (size_t)n * (2 * c) + head * c;
  const float* sp = attS + head * c;
  const float* dp = attD + head * c;
  float a = 0.f, b = 0.f;
  for (int j = 0; j < c; ++j) { float v = hp[j]; a += v * sp[j]; b += v * dp[j]; }
  asrc[idx] = a;
  adst[idx] = b;
}

__global__ void edge_max(const int* __restrict__ src, const int* __restrict__ dst,
                         const float* __restrict__ asrc, const float* __restrict__ adst,
                         float* __restrict__ ebuf, float* __restrict__ emax, int E) {
  int e = blockIdx.x * blockDim.x + threadIdx.x;
  if (e >= E) return;
  int s = src[e], d = dst[e];
#pragma unroll
  for (int hh = 0; hh < 2; ++hh) {
    float v = asrc[s * 2 + hh] + adst[d * 2 + hh];
    v = (v > 0.f) ? v : 0.2f * v;             // leaky_relu, slope 0.2
    ebuf[e * 2 + hh] = v;
    atomicMaxF(&emax[d * 2 + hh], v);
  }
}

__global__ void edge_exp(const int* __restrict__ dst, float* __restrict__ ebuf,
                         const float* __restrict__ emax, float* __restrict__ den, int E) {
  int i = blockIdx.x * blockDim.x + threadIdx.x;
  if (i >= E * 2) return;
  int e = i >> 1, hh = i & 1;
  int d = dst[e];
  float z = __expf(ebuf[i] - emax[d * 2 + hh]);
  ebuf[i] = z;
  atomicAdd(&den[d * 2 + hh], z);
}

// One thread per (edge, feature): coalesced gather of h[src], f32 L2 atomic scatter to agg[dst].
__global__ void edge_aggr(const int* __restrict__ src, const int* __restrict__ dst,
                          const float* __restrict__ h, const float* __restrict__ ez,
                          const float* __restrict__ den, float* __restrict__ agg,
                          int E, int hcShift, int cShift) {
  int i = blockIdx.x * blockDim.x + threadIdx.x;
  if (i >= (E << hcShift)) return;
  int e = i >> hcShift;
  int f = i & ((1 << hcShift) - 1);
  int head = f >> cShift;
  int s = src[e], d = dst[e];
  float alpha = ez[e * 2 + head] / (den[d * 2 + head] + 1e-16f);
  atomicAdd(&agg[((size_t)d << hcShift) + f], h[((size_t)s << hcShift) + f] * alpha);
}

// Per-channel sum / sumsq over 64-row chunks (thread = channel -> coalesced rows).
__global__ void bn_stats(const float* __restrict__ agg, float* __restrict__ sums,
                         float* __restrict__ sumsq, int Nr, int HC) {
  int ch = threadIdx.x;
  int r0 = blockIdx.x * 64;
  int r1 = min(r0 + 64, Nr);
  float s = 0.f, s2 = 0.f;
  for (int r = r0; r < r1; ++r) {
    float v = agg[(size_t)r * HC + ch];
    s += v; s2 += v * v;
  }
  atomicAdd(&sums[ch], s);
  atomicAdd(&sumsq[ch], s2);
}

// In-place BN(train stats)+ReLU; also emits bf16 copy for the next layer's GEMM A-operand.
__global__ void bn_apply(const float* __restrict__ in, const float* __restrict__ sums,
                         const float* __restrict__ sumsq, const float* __restrict__ gamma,
                         const float* __restrict__ beta, float* __restrict__ y,
                         unsigned short* __restrict__ xb_next, int Nr, int hcShift) {
  int i = blockIdx.x * blockDim.x + threadIdx.x;
  if (i >= (Nr << hcShift)) return;
  int f = i & ((1 << hcShift) - 1);
  float invN = 1.0f / (float)Nr;
  float mu  = sums[f] * invN;
  float var = sumsq[f] * invN - mu * mu;
  float v = (in[i] - mu) * rsqrtf(var + 1e-5f) * gamma[f] + beta[f];
  v = fmaxf(v, 0.f);
  y[i] = v;
  xb_next[i] = f2bf(v);
}

__global__ void pool_max(const float* __restrict__ y, const int* __restrict__ batch,
                         float* __restrict__ pool, int Nr) {
  int i = blockIdx.x * blockDim.x + threadIdx.x;
  if (i >= (Nr << 7)) return;
  int n = i >> 7, f = i & 127;
  atomicMaxF(&pool[(batch[n] << 7) + f], y[i]);
}

__global__ void mlp_head(const float* __restrict__ pool, const float* __restrict__ W1,
                         const float* __restrict__ b1, const float* __restrict__ W2,
                         const float* __restrict__ b2, float* __restrict__ out) {
  __shared__ float hid[16 * 64];
  for (int t = threadIdx.x; t < 16 * 64; t += blockDim.x) {
    int g = t >> 6, j = t & 63;
    float acc = b1[j];
    for (int k = 0; k < 128; ++k) {
      float gv = pool[g * 128 + k];
      if (!__builtin_isfinite(gv)) gv = 0.f;
      acc += gv * W1[k * 64 + j];
    }
    hid[t] = fmaxf(acc, 0.f);
  }
  __syncthreads();
  if (threadIdx.x < 16) {
    float acc = b2[0];
    for (int k = 0; k < 64; ++k) acc += hid[threadIdx.x * 64 + k] * W2[k];
    out[threadIdx.x] = acc;
  }
}

extern "C" void kernel_launch(void* const* d_in, const int* in_sizes, int n_in,
                              void* d_out, int out_size, void* d_ws, size_t ws_size,
                              hipStream_t stream) {
  (void)n_in; (void)out_size; (void)ws_size;
  const float* x   = (const float*)d_in[0];
  const int*   ei  = (const int*)d_in[1];
  const int*   bat = (const int*)d_in[2];
  const int N = in_sizes[0] / 128;
  const int E = in_sizes[1] / 2;
  const int* srcI = ei;
  const int* dstI = ei + E;

  const float *Wl[3], *attS[3], *attD[3], *gam[3], *bet[3];
  for (int l = 0; l < 3; ++l) {
    int b = 3 + 6 * l;                    // W, att_src, att_dst, bias, gamma, beta
    Wl[l]   = (const float*)d_in[b + 0];
    attS[l] = (const float*)d_in[b + 1];
    attD[l] = (const float*)d_in[b + 2];
    gam[l]  = (const float*)d_in[b + 4];
    bet[l]  = (const float*)d_in[b + 5];
  }
  const float* W1 = (const float*)d_in[21];
  const float* b1 = (const float*)d_in[22];
  const float* W2 = (const float*)d_in[23];
  const float* b2 = (const float*)d_in[24];
  float* outp = (float*)d_out;

  size_t off = 0;
  auto carve = [&](size_t bytes) -> void* {
    void* p = (char*)d_ws + off;
    off += (bytes + 255) & ~(size_t)255;
    return p;
  };
  unsigned short* xb = (unsigned short*)carve((size_t)N * 256 * 2);
  unsigned short* wt = (unsigned short*)carve((size_t)256 * 256 * 2);
  float* hbuf  = (float*)carve((size_t)N * 256 * 4);
  float* agg   = (float*)carve((size_t)N * 256 * 4);
  float* asrc  = (float*)carve((size_t)N * 2 * 4);
  float* adst  = (float*)carve((size_t)N * 2 * 4);
  float* emax  = (float*)carve((size_t)N * 2 * 4);
  float* den   = (float*)carve((size_t)N * 2 * 4);
  float* ebuf  = (float*)carve((size_t)E * 2 * 4);
  float* sums  = (float*)carve(256 * 4);
  float* sumsq = (float*)carve(256 * 4);
  float* pool  = (float*)carve(16 * 128 * 4);

  const int Cin[3]  = {128, 128, 256};
  const int cdim[3] = {64, 128, 64};

  cvt_f32_bf16<<<dim3((N * 128 + 255) / 256), dim3(256), 0, stream>>>(x, xb, N * 128);

  for (int l = 0; l < 3; ++l) {
    int K = Cin[l], c_ = cdim[l], HC = 2 * c_;
    int hcShift = (HC == 256) ? 8 : 7;
    int cShift  = (c_ == 128) ? 7 : 6;

    cvt_w_t_bf16<<<dim3((K * HC + 255) / 256), dim3(256), 0, stream>>>(Wl[l], wt, K, HC, hcShift);

    int nRowTiles = N / 16;
    gemm_bf16_wmma<<<dim3((nRowTiles + 3) / 4, HC / 16), dim3(128), 0, stream>>>(
        xb, wt, hbuf, nRowTiles, K, HC);

    attn_node<<<dim3((N * 2 + 255) / 256), dim3(256), 0, stream>>>(
        hbuf, attS[l], attD[l], asrc, adst, N, c_);

    fill_f32<<<dim3((N * 2 + 255) / 256), dim3(256), 0, stream>>>(emax, -INFINITY, N * 2);
    fill_f32<<<dim3((N * 2 + 255) / 256), dim3(256), 0, stream>>>(den, 0.f, N * 2);
    fill_f32<<<dim3((N * HC + 255) / 256), dim3(256), 0, stream>>>(agg, 0.f, N * HC);
    fill_f32<<<dim3(1), dim3(256), 0, stream>>>(sums, 0.f, HC);
    fill_f32<<<dim3(1), dim3(256), 0, stream>>>(sumsq, 0.f, HC);

    edge_max<<<dim3((E + 255) / 256), dim3(256), 0, stream>>>(srcI, dstI, asrc, adst, ebuf, emax, E);
    edge_exp<<<dim3((E * 2 + 255) / 256), dim3(256), 0, stream>>>(dstI, ebuf, emax, den, E);
    int tot = E << hcShift;
    edge_aggr<<<dim3((tot + 255) / 256), dim3(256), 0, stream>>>(
        srcI, dstI, hbuf, ebuf, den, agg, E, hcShift, cShift);

    bn_stats<<<dim3((N + 63) / 64), dim3(HC), 0, stream>>>(agg, sums, sumsq, N, HC);
    bn_apply<<<dim3(((N << hcShift) + 255) / 256), dim3(256), 0, stream>>>(
        agg, sums, sumsq, gam[l], bet[l], agg, xb, N, hcShift);
  }

  fill_f32<<<dim3((16 * 128 + 255) / 256), dim3(256), 0, stream>>>(pool, -INFINITY, 16 * 128);
  pool_max<<<dim3(((N << 7) + 255) / 256), dim3(256), 0, stream>>>(agg, bat, pool, N);
  mlp_head<<<dim3(1), dim3(256), 0, stream>>>(pool, W1, b1, W2, b2, outp);
}
